// GIN0_25907242729992
// MI455X (gfx1250) — compile-verified, matrix operand order checked by reading, not run on hardware
//
#include <hip/hip_runtime.h>
#include <hip/hip_bf16.h>
#include <math.h>

// ---------------- problem constants (match reference) ----------------
#define NN   50000     // nodes  (= 3125 * 16 row tiles)
#define NE   625000    // edges
#define NG   256       // graphs (= 16 * 16)
#define CH   128       // channels
#define NL   3         // conv layers
#define NOUT 10        // classes

typedef __attribute__((ext_vector_type(16))) __bf16 v16bf;
typedef __attribute__((ext_vector_type(8)))  float  v8f;

// ISA 7.12.2 16-bit A/B fragment K-pattern for lane `ln`, element `e`:
//   khalf = (ln>=16) ? 8 : 0 ; pairs 0..3 -> K=khalf+{0..7}, pairs 4..7 -> K=khalf+{16..23}
__device__ __forceinline__ int frag_k(int kt, int ln, int e) {
    int p = e >> 1;
    return kt * 32 + ((ln >> 4) << 3) + 2 * (p & 3) + ((p & 4) ? 16 : 0) + (e & 1);
}

// ---------------- weight pre-pack: f32 [128x128] -> bf16 fragment-major ----
// Wp layout: idx = ((nt*4 + kt)*32 + lane)*16 + e  (nt = 16-col tile 0..7)
// so each (wave,kt) B fragment is one contiguous 32B per lane -> 2x b128 load.
__global__ void pack_w_kernel(const float* __restrict__ W,
                              __bf16* __restrict__ Wp, int nmats) {
    int i = blockIdx.x * blockDim.x + threadIdx.x;
    if (i >= nmats * CH * CH) return;
    int mat = i >> 14;
    int j   = i & 16383;
    int e   = j & 15;
    int ln  = (j >> 4) & 31;
    int kt  = (j >> 9) & 3;
    int nt  = j >> 11;
    int k   = frag_k(kt, ln, e);
    int n   = nt * 16 + (ln & 15);
    Wp[i] = (__bf16)W[(size_t)mat * CH * CH + (size_t)k * CH + n];
}

// ---------------- small utility kernels ----------------
__global__ void copy_f4_kernel(const float4* __restrict__ in,
                               float4* __restrict__ out, int n4) {
    int i = blockIdx.x * blockDim.x + threadIdx.x;
    if (i < n4) out[i] = in[i];
}

__global__ void zero_f_kernel(float* __restrict__ p, int n) {
    int i = blockIdx.x * blockDim.x + threadIdx.x;
    if (i < n) p[i] = 0.0f;
}

// ---------------- edge scatter-add: z[dst] += h[src] ----------------
// one wave32 per edge: lane l moves channels [4l,4l+3] (float4 gather,
// 4x global_atomic_add_f32 scatter). Node tensor (25.6MB) resolves in L2.
__global__ void scatter_add_edges_kernel(const float* __restrict__ h,
                                         const int* __restrict__ esrc,
                                         const int* __restrict__ edst,
                                         float* __restrict__ z, int ne) {
    unsigned tid = blockIdx.x * blockDim.x + threadIdx.x;
    unsigned e   = tid >> 5;
    if (e >= (unsigned)ne) return;
    int lane = tid & 31;
    int s = esrc[e];
    int d = edst[e];
    const float4 v = ((const float4*)(h + (size_t)s * CH))[lane];
    float* dp = z + (size_t)d * CH + lane * 4;
    atomicAdd(dp + 0, v.x);
    atomicAdd(dp + 1, v.y);
    atomicAdd(dp + 2, v.z);
    atomicAdd(dp + 3, v.w);
}

// ---------------- WMMA GEMM: C[M x 128] = act(A @ W + b), K = N = 128 -------
// block = 256 thr = 8 waves; block owns 64 rows (4 row tiles), wave w owns
// 16-col tile n0 = 16w. B fragments (all K) held in registers for the whole
// block -> 16 v_wmma_f32_16x16x32_bf16 per wave. A tiles staged in LDS in
// fragment-major order (contiguous 32B/lane -> ds_load_b128 pairs).
__global__ void gemm128_bf16_kernel(const float* __restrict__ A,
                                    const __bf16* __restrict__ Wp,
                                    const float* __restrict__ bias,
                                    float* __restrict__ Cout, int M, int relu) {
    __shared__ __align__(32) __bf16 Af[4 * 2048];   // 4 tiles x (4kt x 32ln x 16e)

    const int lane   = threadIdx.x & 31;
    const int wave   = threadIdx.x >> 5;            // 0..7
    const int m_base = blockIdx.x * 64;
    const int lrow   = lane & 15;
    const int khalf  = (lane >> 4) << 3;

    // stage 4 row tiles of A (f32 -> bf16) directly in fragment order
    for (int i = threadIdx.x; i < 4 * 2048; i += 256) {
        int rt = i >> 11;
        int j  = i & 2047;
        int e  = j & 15;
        int ln = (j >> 4) & 31;
        int kt = j >> 9;
        int k  = frag_k(kt, ln, e);
        int m  = m_base + rt * 16 + (ln & 15);
        Af[i] = (m < M) ? (__bf16)A[(size_t)m * CH + k] : (__bf16)0.0f;
    }
    __syncthreads();

    // this wave's B fragments for all of K (4 x 32B per lane, b128 pairs)
    const __bf16* wb = Wp + ((size_t)wave * 4 * 32 + lane) * 16;
    const v16bf b0 = *(const v16bf*)(wb + 0 * 512);
    const v16bf b1 = *(const v16bf*)(wb + 1 * 512);
    const v16bf b2 = *(const v16bf*)(wb + 2 * 512);
    const v16bf b3 = *(const v16bf*)(wb + 3 * 512);

    const int   n  = wave * 16 + lrow;
    const float bv = bias[n];

#pragma unroll
    for (int rt = 0; rt < 4; ++rt) {
        const int m0 = m_base + rt * 16;
        if (m0 < M) {
            const __bf16* af = Af + rt * 2048 + (size_t)lane * 16;
            const v16bf a0 = *(const v16bf*)(af + 0 * 512);
            const v16bf a1 = *(const v16bf*)(af + 1 * 512);
            const v16bf a2 = *(const v16bf*)(af + 2 * 512);
            const v16bf a3 = *(const v16bf*)(af + 3 * 512);

            v8f acc = {};
            acc = __builtin_amdgcn_wmma_f32_16x16x32_bf16(false, a0, false, b0,
                                                          (short)0, acc, false, false);
            acc = __builtin_amdgcn_wmma_f32_16x16x32_bf16(false, a1, false, b1,
                                                          (short)0, acc, false, false);
            acc = __builtin_amdgcn_wmma_f32_16x16x32_bf16(false, a2, false, b2,
                                                          (short)0, acc, false, false);
            acc = __builtin_amdgcn_wmma_f32_16x16x32_bf16(false, a3, false, b3,
                                                          (short)0, acc, false, false);

            // epilogue: bias + optional ReLU (C/D layout: VGPR v -> row v / v+8)
#pragma unroll
            for (int v = 0; v < 8; ++v) {
                float val = acc[v] + bv;
                if (relu) val = fmaxf(val, 0.0f);
                Cout[(size_t)(m0 + v + khalf) * CH + n] = val;
            }
        }
    }
}

// ---------------- graph mean-pool scatter ----------------
__global__ void pool_scatter_kernel(const float* __restrict__ h,
                                    const int* __restrict__ gid,
                                    float* __restrict__ sums,
                                    float* __restrict__ cnt, int nn) {
    unsigned tid = blockIdx.x * blockDim.x + threadIdx.x;
    unsigned n   = tid >> 5;
    if (n >= (unsigned)nn) return;
    int lane = tid & 31;
    int g    = gid[n];
    const float4 v = ((const float4*)(h + (size_t)n * CH))[lane];
    float* dp = sums + (size_t)g * CH + lane * 4;
    atomicAdd(dp + 0, v.x);
    atomicAdd(dp + 1, v.y);
    atomicAdd(dp + 2, v.z);
    atomicAdd(dp + 3, v.w);
    if (lane == 0) atomicAdd(&cnt[g], 1.0f);
}

__global__ void pool_div_kernel(float* __restrict__ sums,
                                const float* __restrict__ cnt, int n) {
    int i = blockIdx.x * blockDim.x + threadIdx.x;
    if (i < n) sums[i] /= fmaxf(cnt[i >> 7], 1.0f);
}

// ---------------- head: logits (128 x 10) + softmax, one thread/graph ------
__global__ void head_softmax_kernel(const float* __restrict__ p2,
                                    const float* __restrict__ W,
                                    const float* __restrict__ b,
                                    float* __restrict__ out) {
    int g = blockIdx.x * blockDim.x + threadIdx.x;
    if (g >= NG) return;
    float logits[NOUT];
#pragma unroll
    for (int o = 0; o < NOUT; ++o) {
        float s = b[o];
        for (int k = 0; k < CH; ++k) s += p2[(size_t)g * CH + k] * W[k * NOUT + o];
        logits[o] = s;
    }
    float mx = logits[0];
#pragma unroll
    for (int o = 1; o < NOUT; ++o) mx = fmaxf(mx, logits[o]);
    float sum = 0.0f;
#pragma unroll
    for (int o = 0; o < NOUT; ++o) { logits[o] = expf(logits[o] - mx); sum += logits[o]; }
    float inv = 1.0f / sum;
#pragma unroll
    for (int o = 0; o < NOUT; ++o) out[(size_t)g * NOUT + o] = logits[o] * inv;
}

// ---------------- launcher ----------------
extern "C" void kernel_launch(void* const* d_in, const int* in_sizes, int n_in,
                              void* d_out, int out_size, void* d_ws, size_t ws_size,
                              hipStream_t stream) {
    (void)in_sizes; (void)n_in; (void)out_size; (void)ws_size;

    const float* x     = (const float*)d_in[0];
    const int*   esrc  = (const int*)d_in[1];
    const int*   edst  = (const int*)d_in[2];
    const int*   gid   = (const int*)d_in[3];
    const float* convW = (const float*)d_in[4];   // (3,3,128,128)
    const float* convB = (const float*)d_in[5];   // (3,3,128)
    const float* d1W   = (const float*)d_in[6];   // (128,128)
    const float* d1b   = (const float*)d_in[7];   // (128,)
    const float* d2W   = (const float*)d_in[8];   // (128,10)
    const float* d2b   = (const float*)d_in[9];   // (10,)
    float* out = (float*)d_out;                   // (256,10) f32

    // workspace carve-up (~52 MB; ping-pong activations stay inside 192MB L2)
    float* bufB   = (float*)d_ws;                 // NN*CH
    float* bufC   = bufB + (size_t)NN * CH;       // NN*CH
    float* pooled = bufC + (size_t)NN * CH;       // NG*CH
    float* p2     = pooled + (size_t)NG * CH;     // NG*CH
    float* cnt    = p2 + (size_t)NG * CH;         // NG
    __bf16* wbf   = (__bf16*)(cnt + NG);          // 10 x 128x128 bf16, frag-major

    const int WMAT = CH * CH;                     // 16384
    const int TPB  = 256;

    // 1) one-time weight conversion + fragment-order shuffle (f32 -> bf16)
    pack_w_kernel<<<(9 * WMAT + TPB - 1) / TPB, TPB, 0, stream>>>(convW, wbf, 9);
    pack_w_kernel<<<(WMAT + TPB - 1) / TPB, TPB, 0, stream>>>(d1W, wbf + (size_t)9 * WMAT, 1);

    const int nF4      = NN * CH / 4;
    const int gridCopy = (nF4 + TPB - 1) / TPB;
    const int gridScat = (int)(((size_t)NE * 32 + TPB - 1) / TPB);
    const int gridGemm = (NN / 16 + 3) / 4;       // 782 blocks x 64 rows

    // 2) three GIN conv layers
    const float* cur = x;
    for (int l = 0; l < NL; ++l) {
        // z = h (copy) then z += A@h (edge scatter-add) => z = h + agg
        copy_f4_kernel<<<gridCopy, TPB, 0, stream>>>((const float4*)cur, (float4*)bufB, nF4);
        scatter_add_edges_kernel<<<gridScat, TPB, 0, stream>>>(cur, esrc, edst, bufB, NE);
        // MLP: Dense+ReLU, Dense+ReLU, Dense (bias+act fused into WMMA epilogue)
        gemm128_bf16_kernel<<<gridGemm, TPB, 0, stream>>>(
            bufB, wbf + (size_t)(l * 3 + 0) * WMAT, convB + (l * 3 + 0) * CH, bufC, NN, 1);
        gemm128_bf16_kernel<<<gridGemm, TPB, 0, stream>>>(
            bufC, wbf + (size_t)(l * 3 + 1) * WMAT, convB + (l * 3 + 1) * CH, bufB, NN, 1);
        gemm128_bf16_kernel<<<gridGemm, TPB, 0, stream>>>(
            bufB, wbf + (size_t)(l * 3 + 2) * WMAT, convB + (l * 3 + 2) * CH, bufC, NN, 0);
        cur = bufC;
    }

    // 3) segment mean pool over graphs
    const int nPoolZero = 2 * NG * CH + NG;       // pooled + p2 + cnt (contiguous)
    zero_f_kernel<<<(nPoolZero + TPB - 1) / TPB, TPB, 0, stream>>>(pooled, nPoolZero);
    pool_scatter_kernel<<<(int)(((size_t)NN * 32 + TPB - 1) / TPB), TPB, 0, stream>>>(
        cur, gid, pooled, cnt, NN);
    pool_div_kernel<<<(NG * CH + TPB - 1) / TPB, TPB, 0, stream>>>(pooled, cnt, NG * CH);

    // 4) head: Dense+ReLU (WMMA, M=256 = 4 blocks) then logits + softmax
    gemm128_bf16_kernel<<<NG / 64, TPB, 0, stream>>>(
        pooled, wbf + (size_t)9 * WMAT, d1b, p2, NG, 1);
    head_softmax_kernel<<<1, NG, 0, stream>>>(p2, d2W, d2b, out);
}
